// MultiHeadSelfAttention_89678917141241
// MI455X (gfx1250) — matrix operand order repaired and checked by reading, not compiled, not run on hardware
//
#include <hip/hip_runtime.h>
#include <hip/hip_bf16.h>

// Problem constants (B,S,D,H,DK) = (4,2048,1024,16,64)
#define BB 4
#define SS 2048
#define DD 1024
#define HH 16
#define DK 64
#define MT (BB*SS)   // 8192 rows total

typedef __attribute__((ext_vector_type(16))) __bf16 v16bf;
typedef __attribute__((ext_vector_type(8)))  float  v8f;
typedef int v4i32 __attribute__((vector_size(16)));

union FragB16 { v16bf bf; uint4 q[2]; };

__device__ __forceinline__ unsigned short f32_to_bf16(float f) {
    unsigned int u = __float_as_uint(f);
    unsigned int r = u + 0x7FFFu + ((u >> 16) & 1u);   // round-to-nearest-even
    return (unsigned short)(r >> 16);
}

// ---- CDNA5 async global->LDS staging (probe builtins, fall back cleanly) ----
#if __has_builtin(__builtin_amdgcn_global_load_async_to_lds_b128)
#define HAVE_ASYNC_LDS 1
#else
#define HAVE_ASYNC_LDS 0
#endif

#if __has_builtin(__builtin_amdgcn_s_wait_asynccnt)
#define WAIT_ASYNC(n) __builtin_amdgcn_s_wait_asynccnt(n)
#else
#define WAIT_ASYNC(n) asm volatile("s_wait_asynccnt %0" :: "i"(n) : "memory")
#endif

__device__ __forceinline__ void stage16(const unsigned short* g, unsigned short* l) {
#if HAVE_ASYNC_LDS
    __builtin_amdgcn_global_load_async_to_lds_b128(
        (__attribute__((address_space(1))) v4i32*)g,
        (__attribute__((address_space(3))) v4i32*)l,
        0, 0);
#else
    *(uint4*)l = *(const uint4*)g;   // synchronous fallback, same structure
#endif
}

// ---------------------------------------------------------------------------
// fp32 -> bf16 conversion (vectorized, memory bound)
// ---------------------------------------------------------------------------
__global__ __launch_bounds__(256) void cvt_f32_bf16(const float* __restrict__ in,
                                                    unsigned short* __restrict__ out,
                                                    int n4) {
    int i = blockIdx.x * blockDim.x + threadIdx.x;
    if (i < n4) {
        float4 f = reinterpret_cast<const float4*>(in)[i];
        ushort4 o;
        o.x = f32_to_bf16(f.x); o.y = f32_to_bf16(f.y);
        o.z = f32_to_bf16(f.z); o.w = f32_to_bf16(f.w);
        reinterpret_cast<ushort4*>(out)[i] = o;
    }
}

// ---------------------------------------------------------------------------
// bf16 GEMM: C[m,n] = sum_k A[m,k]*W[n,k] + bias[n]      (M=8192,N=1024,K=1024)
// 8 waves/block, each wave does a 16(M) x 64(N) strip via 4 f32 accumulators.
// MODE 0: store bf16 to (B,H,S,DK), scaled            (Q,K projections)
// MODE 1: store fp32 to (M,N) row-major               (final output projection)
// MODE 2: store bf16 to (B,H,DK,S)  == V pre-transposed per head (free here,
//         makes P@V B-fragments contiguous in the attention kernel)
// ---------------------------------------------------------------------------
template<int MODE>
__global__ __launch_bounds__(256) void gemm_bf16(const unsigned short* __restrict__ A,
                                                 const unsigned short* __restrict__ W,
                                                 const float* __restrict__ bias,
                                                 unsigned short* __restrict__ outb,
                                                 float* __restrict__ outf,
                                                 float scale) {
    const int wave  = threadIdx.x >> 5;
    const int lane  = threadIdx.x & 31;
    const int half  = lane >> 4;
    const int l16   = lane & 15;
    const int mbase = blockIdx.x * 128 + wave * 16;
    const int nbase = blockIdx.y * 64;

    v8f acc[4] = {};
    const unsigned short* arow = A + (size_t)(mbase + l16) * DD;

    for (int k0 = 0; k0 < DD; k0 += 32) {
        FragB16 a;
        a.q[0] = *(const uint4*)(arow + k0 + half * 8);
        a.q[1] = *(const uint4*)(arow + k0 + 16 + half * 8);
        __builtin_prefetch(arow + k0 + 128, 0, 0);          // global_prefetch ahead
#pragma unroll
        for (int t = 0; t < 4; ++t) {
            const unsigned short* wrow =
                W + (size_t)(nbase + t * 16 + l16) * DD + k0 + half * 8;
            FragB16 b;
            b.q[0] = *(const uint4*)(wrow);
            b.q[1] = *(const uint4*)(wrow + 16);
            acc[t] = __builtin_amdgcn_wmma_f32_16x16x32_bf16(
                false, a.bf, false, b.bf, (short)0, acc[t], false, false);
        }
    }

#pragma unroll
    for (int t = 0; t < 4; ++t) {
#pragma unroll
        for (int r = 0; r < 8; ++r) {
            const int m = mbase + r + half * 8;
            const int n = nbase + t * 16 + l16;
            float v = acc[t][r] + bias[n];
            if (MODE == 0) {
                const int b  = m >> 11, s = m & (SS - 1);
                const int h  = n >> 6,  dk = n & (DK - 1);
                outb[((((size_t)b * HH + h) * SS) + s) * DK + dk] = f32_to_bf16(v * scale);
            } else if (MODE == 2) {
                const int b  = m >> 11, s = m & (SS - 1);
                const int h  = n >> 6,  dk = n & (DK - 1);
                outb[((((size_t)b * HH + h) * DK) + dk) * SS + s] = f32_to_bf16(v);
            } else {
                outf[(size_t)m * DD + n] = v;
            }
        }
    }
}

// ---------------------------------------------------------------------------
// Flash attention per (b,h). Q pre-scaled by 1/sqrt(DK) at projection time.
// Block = 8 waves; wave handles 16 q-rows; keys processed 32 at a time.
// K (row-major S x DK) and Vt (DK x S) chunks are async-staged into
// double-buffered LDS (shared by all 8 waves); fragments then come from
// contiguous ds_load_b128.  exp(P) bounces through wave-private LDS to
// convert C-layout -> A-fragment layout (no barrier needed).
// ---------------------------------------------------------------------------
__global__ __launch_bounds__(256) void flash_attn(const unsigned short* __restrict__ Q,
                                                  const unsigned short* __restrict__ K,
                                                  const unsigned short* __restrict__ Vt,
                                                  unsigned short* __restrict__ Ob) {
    const int bh   = blockIdx.x;               // 0..63
    const int tid  = threadIdx.x;
    const int wave = tid >> 5;
    const int lane = tid & 31;
    const int half = lane >> 4;
    const int l16  = lane & 15;
    const int qrow = blockIdx.y * 128 + wave * 16;
    const size_t headoff = (size_t)bh * SS * DK;

    __shared__ alignas(16) unsigned short p_lds[8 * 16 * 32];   // per-wave P tiles (8 KB)
    __shared__ alignas(16) unsigned short kbuf[2][32 * DK];     // K chunks      (8 KB)
    __shared__ alignas(16) unsigned short vbuf[2][DK * 32];     // Vt chunks     (8 KB)

    // staging coordinates (one b128 per thread per buffer)
    const int krow = tid >> 3, kc0 = (tid & 7) * 8;   // K  : [key 0..31][dk]
    const int vrow = tid >> 2, vc0 = (tid & 3) * 8;   // Vt : [dk 0..63][key]

    // Q A-fragments for DK 0..31 and 32..63 (loaded once from global)
    const unsigned short* qr = Q + headoff + (size_t)(qrow + l16) * DK;
    FragB16 a0, a1;
    a0.q[0] = *(const uint4*)(qr + half * 8);
    a0.q[1] = *(const uint4*)(qr + 16 + half * 8);
    a1.q[0] = *(const uint4*)(qr + 32 + half * 8);
    a1.q[1] = *(const uint4*)(qr + 48 + half * 8);

    float mrow[8], lrow[8];
    v8f o[4] = {};
#pragma unroll
    for (int r = 0; r < 8; ++r) { mrow[r] = -1.0e30f; lrow[r] = 0.0f; }

    unsigned short* myp = p_lds + wave * 512;

    // prologue: stage chunk 0
    stage16(K  + headoff + (size_t)krow * DK + kc0, &kbuf[0][krow * DK + kc0]);
    stage16(Vt + headoff + (size_t)vrow * SS + vc0, &vbuf[0][vrow * 32 + vc0]);

    for (int jt = 0; jt < SS / 32; ++jt) {
        const int cur = jt & 1;
        if (jt + 1 < SS / 32) {
            const int j1 = (jt + 1) * 32;
            stage16(K  + headoff + (size_t)(j1 + krow) * DK + kc0,
                    &kbuf[cur ^ 1][krow * DK + kc0]);
            stage16(Vt + headoff + (size_t)vrow * SS + j1 + vc0,
                    &vbuf[cur ^ 1][vrow * 32 + vc0]);
            WAIT_ASYNC(2);          // current chunk done; next chunk stays in flight
        } else {
            WAIT_ASYNC(0);
        }
        __syncthreads();            // all waves' staging of `cur` visible

        // scores for 32 keys = two 16x16 C tiles (B frags contiguous in kbuf)
        v8f sc[2] = {};
#pragma unroll
        for (int t = 0; t < 2; ++t) {
            const unsigned short* krp = &kbuf[cur][(t * 16 + l16) * DK + half * 8];
            FragB16 b0, b1;
            b0.q[0] = *(const uint4*)(krp);
            b0.q[1] = *(const uint4*)(krp + 16);
            b1.q[0] = *(const uint4*)(krp + 32);
            b1.q[1] = *(const uint4*)(krp + 48);
            sc[t] = __builtin_amdgcn_wmma_f32_16x16x32_bf16(
                false, a0.bf, false, b0.bf, (short)0, sc[t], false, false);
            sc[t] = __builtin_amdgcn_wmma_f32_16x16x32_bf16(
                false, a1.bf, false, b1.bf, (short)0, sc[t], false, false);
        }

        // online softmax: row stats per-VGPR, replicated across 16-lane halves
        float alpha[8];
#pragma unroll
        for (int r = 0; r < 8; ++r) {
            float v = fmaxf(sc[0][r], sc[1][r]);
            v = fmaxf(v, __shfl_xor(v, 1));
            v = fmaxf(v, __shfl_xor(v, 2));
            v = fmaxf(v, __shfl_xor(v, 4));
            v = fmaxf(v, __shfl_xor(v, 8));
            const float mnew = fmaxf(mrow[r], v);
            alpha[r] = __expf(mrow[r] - mnew);
            mrow[r]  = mnew;
        }
#pragma unroll
        for (int r = 0; r < 8; ++r) {
            const float p0 = __expf(sc[0][r] - mrow[r]);
            const float p1 = __expf(sc[1][r] - mrow[r]);
            const int row = r + half * 8;
            myp[row * 32 + l16]      = f32_to_bf16(p0);
            myp[row * 32 + 16 + l16] = f32_to_bf16(p1);
            float s = p0 + p1;
            s += __shfl_xor(s, 1);
            s += __shfl_xor(s, 2);
            s += __shfl_xor(s, 4);
            s += __shfl_xor(s, 8);
            lrow[r] = lrow[r] * alpha[r] + s;
        }
        asm volatile("s_wait_dscnt 0" ::: "memory");   // our P stores -> our loads

        // reload P as A-fragment (wave-private, no barrier)
        FragB16 pf;
        const unsigned short* pr = myp + l16 * 32;
        pf.q[0] = *(const uint4*)(pr + half * 8);
        pf.q[1] = *(const uint4*)(pr + 16 + half * 8);

        // rescale O and accumulate P@V (B frags contiguous in vbuf)
#pragma unroll
        for (int t = 0; t < 4; ++t) {
#pragma unroll
            for (int r = 0; r < 8; ++r) o[t][r] *= alpha[r];
            FragB16 vb;
            const unsigned short* vtp = &vbuf[cur][(t * 16 + l16) * 32 + half * 8];
            vb.q[0] = *(const uint4*)(vtp);
            vb.q[1] = *(const uint4*)(vtp + 16);
            o[t] = __builtin_amdgcn_wmma_f32_16x16x32_bf16(
                false, pf.bf, false, vb.bf, (short)0, o[t], false, false);
        }
        __syncthreads();            // everyone done reading `cur` before restaging it
    }

    // normalize and store attn output as bf16 in (B,S,D) layout
    const int b = bh >> 4, h = bh & 15;
#pragma unroll
    for (int t = 0; t < 4; ++t) {
#pragma unroll
        for (int r = 0; r < 8; ++r) {
            const int s = qrow + r + half * 8;
            const int d = h * DK + t * 16 + l16;
            Ob[((size_t)b * SS + s) * DD + d] = f32_to_bf16(o[t][r] / lrow[r]);
        }
    }
}

// ---------------------------------------------------------------------------
extern "C" void kernel_launch(void* const* d_in, const int* in_sizes, int n_in,
                              void* d_out, int out_size, void* d_ws, size_t ws_size,
                              hipStream_t stream) {
    const float* x  = (const float*)d_in[0];
    const float* Wq = (const float*)d_in[1];
    const float* bq = (const float*)d_in[2];
    const float* Wk = (const float*)d_in[3];
    const float* bk = (const float*)d_in[4];
    const float* Wv = (const float*)d_in[5];
    const float* bv = (const float*)d_in[6];
    const float* Wo = (const float*)d_in[7];
    const float* bo = (const float*)d_in[8];
    float* out = (float*)d_out;

    char* ws = (char*)d_ws;
    unsigned short* Xb  = (unsigned short*)(ws);                       // 16 MiB
    unsigned short* Wqb = (unsigned short*)(ws + (size_t)16777216);    // 2 MiB each
    unsigned short* Wkb = Wqb + 1048576;
    unsigned short* Wvb = Wkb + 1048576;
    unsigned short* Wob = Wvb + 1048576;
    unsigned short* Qw  = (unsigned short*)(ws + (size_t)16777216 + 8388608);
    unsigned short* Kw  = Qw + (size_t)MT * DD;
    unsigned short* Vw  = Kw + (size_t)MT * DD;   // V stored (B,H,DK,S)
    unsigned short* Aw  = Vw + (size_t)MT * DD;

    // 1) fp32 -> bf16
    {
        const int nx4 = (MT * DD) / 4;       // 2,097,152
        const int nw4 = (DD * DD) / 4;       //   262,144
        cvt_f32_bf16<<<(nx4 + 255) / 256, 256, 0, stream>>>(x,  Xb,  nx4);
        cvt_f32_bf16<<<(nw4 + 255) / 256, 256, 0, stream>>>(Wq, Wqb, nw4);
        cvt_f32_bf16<<<(nw4 + 255) / 256, 256, 0, stream>>>(Wk, Wkb, nw4);
        cvt_f32_bf16<<<(nw4 + 255) / 256, 256, 0, stream>>>(Wv, Wvb, nw4);
        cvt_f32_bf16<<<(nw4 + 255) / 256, 256, 0, stream>>>(Wo, Wob, nw4);
    }

    // 2) Q/K/V projections (Q carries softmax scale 1/sqrt(64) = 0.125;
    //    V is written pre-transposed per head)
    dim3 ggrid(MT / 128, DD / 64);           // (64, 16)
    gemm_bf16<0><<<ggrid, 256, 0, stream>>>(Xb, Wqb, bq, Qw, nullptr, 0.125f);
    gemm_bf16<0><<<ggrid, 256, 0, stream>>>(Xb, Wkb, bk, Kw, nullptr, 1.0f);
    gemm_bf16<2><<<ggrid, 256, 0, stream>>>(Xb, Wvb, bv, Vw, nullptr, 1.0f);

    // 3) attention
    dim3 agrid(BB * HH, SS / 128);           // (64, 16)
    flash_attn<<<agrid, 256, 0, stream>>>(Qw, Kw, Vw, Aw);

    // 4) output projection -> fp32
    gemm_bf16<1><<<ggrid, 256, 0, stream>>>(Aw, Wob, bo, nullptr, out, 1.0f);
}
